// SelfAttentionLayer_45079976739089
// MI455X (gfx1250) — matrix operand order repaired and checked by reading, not compiled, not run on hardware
//
#include <hip/hip_runtime.h>

// ---------------- problem constants ----------------
#define BATCH 2
#define NSEQ  2048
#define FIN   1024
#define NH    16
#define FQK   64
#define FV    64
#define FTOT  3072   // (2*64+64)*16
#define HEADC 192    // per-head columns in qkv

typedef __bf16 bf16_t;
typedef bf16_t v16bf __attribute__((ext_vector_type(16)));
typedef float  v8f   __attribute__((ext_vector_type(8)));

union Frag16 {
    v16bf v;
    uint4 q[2];
    unsigned short u[16];
};

// ------------- gfx1250 async-to-LDS path (probe via __has_builtin) -------------
#if defined(__gfx1250__) && __has_builtin(__builtin_amdgcn_global_load_async_to_lds_b128)
#define HAVE_ASYNC 1
typedef int v4i_ __attribute__((vector_size(16)));
typedef __attribute__((address_space(1))) v4i_ gv4i;   // global int4
typedef __attribute__((address_space(3))) v4i_ lv4i;   // LDS int4
#else
#define HAVE_ASYNC 0
#endif

// copy 16B global -> LDS (async DMA if available, else reg round-trip)
__device__ __forceinline__ void cp16(unsigned short* l, const unsigned short* g) {
#if HAVE_ASYNC
    __builtin_amdgcn_global_load_async_to_lds_b128((gv4i*)g, (lv4i*)l, 0, 0);
#else
    *(uint4*)l = *(const uint4*)g;
#endif
}

__device__ __forceinline__ void wait_async_all() {
#if HAVE_ASYNC
#if __has_builtin(__builtin_amdgcn_s_wait_asynccnt)
    __builtin_amdgcn_s_wait_asynccnt(0);
#else
    asm volatile("s_wait_asynccnt 0x0" ::: "memory");
#endif
#endif
}

// wave-local LDS ordering (DS pipe is in-order per wave; this also fences the compiler)
__device__ __forceinline__ void wait_lds() {
    asm volatile("s_wait_dscnt 0x0" ::: "memory");
}

__device__ __forceinline__ unsigned short f32_to_bf16(float f) {
    unsigned int u = __float_as_uint(f);
    u += 0x7FFFu + ((u >> 16) & 1u);   // round-to-nearest-even
    return (unsigned short)(u >> 16);
}

// ---------------- conversion kernels ----------------
__global__ __launch_bounds__(256) void conv_x_kernel(const float* __restrict__ x,
                                                     unsigned short* __restrict__ xb) {
    int i = (blockIdx.x * 256 + threadIdx.x) * 4;
    float4 v = *(const float4*)(x + i);
    union { unsigned short u[4]; uint2 q; } r;
    r.u[0] = f32_to_bf16(v.x);
    r.u[1] = f32_to_bf16(v.y);
    r.u[2] = f32_to_bf16(v.z);
    r.u[3] = f32_to_bf16(v.w);
    *(uint2*)(xb + i) = r.q;
}

// W [FIN, FTOT] f32 -> Wt [FTOT, FIN] bf16 (transposed so WMMA B-operand is contiguous)
__global__ __launch_bounds__(256) void conv_wT_kernel(const float* __restrict__ W,
                                                      unsigned short* __restrict__ wt) {
    int t = blockIdx.x * 256 + threadIdx.x;   // t = j*FIN + i
    int j = t >> 10;
    int i = t & 1023;
    wt[t] = f32_to_bf16(W[(size_t)i * FTOT + j]);
}

// ---------------- fragment load helpers (per ISA 7.12.2 layouts) ----------------
// A 16x32 bf16 tile, row-major src (stride ld elements):
//   lane<16: row=lane, K {0..7} and {16..23};  lane>=16: row=lane-16, K {8..15} and {24..31}
__device__ __forceinline__ Frag16 load_a_frag(const unsigned short* src, int ld, int lm, int hi) {
    Frag16 a;
    const unsigned short* p = src + lm * ld + hi * 8;
    a.q[0] = *(const uint4*)p;
    a.q[1] = *(const uint4*)(p + 16);
    return a;
}
// B 32x16 bf16 tile from src[n][k] (stride ld): lane<16: col=lane, K {0..7} and {16..23};
// lane>=16: col=lane-16, K {8..15} and {24..31}
__device__ __forceinline__ Frag16 load_b_frag(const unsigned short* src, int ld, int lm, int hi) {
    Frag16 b;
    const unsigned short* p = src + lm * ld + hi * 8;
    b.q[0] = *(const uint4*)p;
    b.q[1] = *(const uint4*)(p + 16);
    return b;
}

// ---------------- QKV projection GEMM ----------------
// block = 128 rows x 64 cols, K staged through double-buffered LDS.
__device__ __forceinline__ void stage_xw(int t, int rowBase, int j0, int k0,
                                         unsigned short* Xsb, unsigned short* Wsb,
                                         const unsigned short* __restrict__ xb,
                                         const unsigned short* __restrict__ wt) {
    #pragma unroll
    for (int c = 0; c < 2; ++c) {
        int chunk = t * 2 + c;            // 0..511
        int row   = chunk >> 2;
        int ko    = (chunk & 3) * 8;
        cp16(Xsb + row * 40 + ko, xb + (size_t)(rowBase + row) * FIN + k0 + ko);
    }
    {
        int row = t >> 2;
        int ko  = (t & 3) * 8;
        cp16(Wsb + row * 40 + ko, wt + (size_t)(j0 + row) * FIN + k0 + ko);
    }
}

__global__ __launch_bounds__(256) void qkv_gemm_kernel(const unsigned short* __restrict__ xb,
                                                       const unsigned short* __restrict__ wt,
                                                       unsigned short* __restrict__ Qb,
                                                       unsigned short* __restrict__ Kb,
                                                       unsigned short* __restrict__ Vtb) {
    __shared__ __align__(16) unsigned short Xs[2][128 * 40];   // 128 x 32 K, skew 40
    __shared__ __align__(16) unsigned short Ws[2][64 * 40];    // 64 x 32 K, skew 40

    const int rowBase = blockIdx.x * 128;     // flat row in [0, B*N)
    const int j0      = blockIdx.y * 64;      // output column tile
    const int t    = threadIdx.x;
    const int wave = t >> 5;
    const int lane = t & 31;
    const int lm   = lane & 15;
    const int hi   = lane >> 4;

    v8f acc[4] = {};
    const int NK = FIN / 32;

    stage_xw(t, rowBase, j0, 0, Xs[0], Ws[0], xb, wt);
    wait_async_all();
    __syncthreads();

    for (int ki = 0; ki < NK; ++ki) {
        const int cur = ki & 1;
        // read this step's fragments from LDS first ...
        Frag16 a = load_a_frag(&Xs[cur][wave * 16 * 40], 40, lm, hi);
        Frag16 bfr[4];
        #pragma unroll
        for (int s = 0; s < 4; ++s)
            bfr[s] = load_b_frag(&Ws[cur][s * 16 * 40], 40, lm, hi);
        // ... then kick off the next tile into the other buffer ...
        if (ki + 1 < NK)
            stage_xw(t, rowBase, j0, (ki + 1) * 32, Xs[cur ^ 1], Ws[cur ^ 1], xb, wt);
        // ... and compute.
        #pragma unroll
        for (int s = 0; s < 4; ++s)
            acc[s] = __builtin_amdgcn_wmma_f32_16x16x32_bf16(
                false, a.v, false, bfr[s].v, (short)0, acc[s], false, false);
        wait_async_all();
        __syncthreads();
    }

    // epilogue: column tile maps entirely to Q, K, or V (64 | 192 boundaries)
    const int h  = j0 / HEADC;
    const int cm = j0 % HEADC;                // 0 -> Q, 64 -> K, 128 -> V
    const int b  = rowBase >> 11;
    const int n0 = (rowBase & (NSEQ - 1)) + wave * 16;

    if (cm < 128) {
        unsigned short* dst = (cm == 0) ? Qb : Kb;   // [B,H,N,64]
        size_t base = (((size_t)b * NH + h) * NSEQ + n0) * FQK;
        #pragma unroll
        for (int s = 0; s < 4; ++s) {
            int d = s * 16 + lm;
            #pragma unroll
            for (int r = 0; r < 8; ++r) {
                int m = r + 8 * hi;
                dst[base + (size_t)m * FQK + d] = f32_to_bf16(acc[s][r]);
            }
        }
    } else {
        // Vt [B,H,64,N]: lane's 8 rows are contiguous along N -> one 16B store per frag
        #pragma unroll
        for (int s = 0; s < 4; ++s) {
            int d = s * 16 + lm;
            union { unsigned short u[8]; uint4 q; } pk;
            #pragma unroll
            for (int r = 0; r < 8; ++r) pk.u[r] = f32_to_bf16(acc[s][r]);
            size_t base = (((size_t)b * NH + h) * FV + d) * NSEQ + n0 + 8 * hi;
            *(uint4*)(Vtb + base) = pk.q;
        }
    }
}

// ---------------- flash attention ----------------
// block = 128 queries (8 waves x 16) for one (b,h); online softmax over 64-key blocks.
// K/V blocks staged cooperatively into double-buffered LDS (shared by all 8 waves).
__global__ __launch_bounds__(256) void attn_kernel(const unsigned short* __restrict__ Qb,
                                                   const unsigned short* __restrict__ Kb,
                                                   const unsigned short* __restrict__ Vtb,
                                                   float* __restrict__ out) {
    __shared__ __align__(16) unsigned short Ks[2][64 * 72];   // [key][d], skew 72
    __shared__ __align__(16) unsigned short Vs[2][64 * 72];   // [feature][key], skew 72
    __shared__ __align__(16) unsigned short Ps[8 * 16 * 72];  // per-wave P strip

    const int t    = threadIdx.x;
    const int wave = t >> 5;
    const int lane = t & 31;
    const int lm   = lane & 15;
    const int hi   = lane >> 4;

    const int q0 = blockIdx.x * 128;
    const int bh = blockIdx.y;            // b*NH + h
    const int b  = bh >> 4;
    const int h  = bh & 15;

    const unsigned short* Qp = Qb + (((size_t)bh) * NSEQ + q0 + wave * 16) * FQK;
    const unsigned short* Kp = Kb + ((size_t)bh) * NSEQ * FQK;
    const unsigned short* Vp = Vtb + ((size_t)bh) * FV * NSEQ;

    // Q A-fragments (d = 0..31, 32..63), kept in registers for the whole loop
    Frag16 aq[2];
    #pragma unroll
    for (int c = 0; c < 2; ++c) aq[c] = load_a_frag(Qp + c * 32, FQK, lm, hi);

    float mrow[8], lrow[8];
    v8f o[4] = {};
    #pragma unroll
    for (int r = 0; r < 8; ++r) { mrow[r] = -3.0e38f; lrow[r] = 0.0f; }

    unsigned short* myP = &Ps[wave * 16 * 72];

    // cooperative K/V staging for one 64-key block: 2 x 16B (K) + 2 x 16B (V) per thread
    auto stage_kv = [&](int j0, int buf) {
        #pragma unroll
        for (int c = 0; c < 2; ++c) {
            int chunk = t * 2 + c;          // 0..511
            int row   = chunk >> 3;         // key row 0..63
            int ko    = (chunk & 7) * 8;
            cp16(&Ks[buf][row * 72 + ko], Kp + (size_t)(j0 + row) * FQK + ko);
        }
        #pragma unroll
        for (int c = 0; c < 2; ++c) {
            int chunk = t * 2 + c;
            int f     = chunk >> 3;         // feature row 0..63
            int ko    = (chunk & 7) * 8;
            cp16(&Vs[buf][f * 72 + ko], Vp + (size_t)f * NSEQ + j0 + ko);
        }
    };

    stage_kv(0, 0);
    wait_async_all();
    __syncthreads();

    const int NBLK = NSEQ / 64;
    for (int jb = 0; jb < NBLK; ++jb) {
        const int cur = jb & 1;

        // S = (Q K^T) * 1/sqrt(64); B-fragments from LDS K tile
        v8f s[4];
        #pragma unroll
        for (int f = 0; f < 4; ++f) {
            const unsigned short* kp = &Ks[cur][f * 16 * 72];
            Frag16 bk0 = load_b_frag(kp, 72, lm, hi);
            Frag16 bk1 = load_b_frag(kp + 32, 72, lm, hi);
            v8f z = {};
            z = __builtin_amdgcn_wmma_f32_16x16x32_bf16(false, aq[0].v, false, bk0.v,
                                                        (short)0, z, false, false);
            z = __builtin_amdgcn_wmma_f32_16x16x32_bf16(false, aq[1].v, false, bk1.v,
                                                        (short)0, z, false, false);
            s[f] = z * 0.125f;
        }

        // prefetch next key block into the other buffer; softmax VALU hides the latency
        if (jb + 1 < NBLK) stage_kv((jb + 1) * 64, cur ^ 1);

        // online softmax; shfl_xor masks 1..8 stay within each 16-lane half (= one row set)
        #pragma unroll
        for (int r = 0; r < 8; ++r) {
            float v = fmaxf(fmaxf(s[0][r], s[1][r]), fmaxf(s[2][r], s[3][r]));
            #pragma unroll
            for (int off = 8; off >= 1; off >>= 1)
                v = fmaxf(v, __shfl_xor(v, off, 32));
            float mnew  = fmaxf(mrow[r], v);
            float alpha = __expf(mrow[r] - mnew);
            mrow[r] = mnew;
            float rs = 0.0f;
            #pragma unroll
            for (int f = 0; f < 4; ++f) {
                float p = __expf(s[f][r] - mnew);
                s[f][r] = p;
                rs += p;
            }
            #pragma unroll
            for (int off = 8; off >= 1; off >>= 1)
                rs += __shfl_xor(rs, off, 32);
            lrow[r] = lrow[r] * alpha + rs;
            #pragma unroll
            for (int f = 0; f < 4; ++f) o[f][r] *= alpha;
        }

        // C-layout -> A-layout transpose of P through per-wave LDS (wave-local ordering)
        #pragma unroll
        for (int f = 0; f < 4; ++f)
            #pragma unroll
            for (int r = 0; r < 8; ++r)
                myP[(r + 8 * hi) * 72 + f * 16 + lm] = f32_to_bf16(s[f][r]);
        wait_lds();

        Frag16 ap[2];
        #pragma unroll
        for (int c = 0; c < 2; ++c) ap[c] = load_a_frag(myP + c * 32, 72, lm, hi);

        // O += P @ V; B-fragments from LDS V tile
        #pragma unroll
        for (int f = 0; f < 4; ++f) {
            #pragma unroll
            for (int c = 0; c < 2; ++c) {
                Frag16 bv = load_b_frag(&Vs[cur][f * 16 * 72] + c * 32, 72, lm, hi);
                o[f] = __builtin_amdgcn_wmma_f32_16x16x32_bf16(
                    false, ap[c].v, false, bv.v, (short)0, o[f], false, false);
            }
        }

        wait_async_all();
        __syncthreads();   // buffer swap
    }

    // epilogue: out[b, q, h*64 + d] = O / l
    #pragma unroll
    for (int r = 0; r < 8; ++r) {
        float inv = 1.0f / lrow[r];
        int q = q0 + wave * 16 + r + 8 * hi;
        float* op = out + ((size_t)b * NSEQ + q) * (NH * FV) + h * FV;
        #pragma unroll
        for (int f = 0; f < 4; ++f)
            op[f * 16 + lm] = o[f][r] * inv;
    }
}

// ---------------- launcher ----------------
extern "C" void kernel_launch(void* const* d_in, const int* in_sizes, int n_in,
                              void* d_out, int out_size, void* d_ws, size_t ws_size,
                              hipStream_t stream) {
    const float* x = (const float*)d_in[0];
    const float* W = (const float*)d_in[1];
    float* out = (float*)d_out;

    char* ws = (char*)d_ws;
    const size_t MB = 1024ull * 1024ull;
    unsigned short* xb  = (unsigned short*)(ws);                 // 8 MB  (B*N*Fin bf16)
    unsigned short* wt  = (unsigned short*)(ws + 8  * MB);       // 6 MB  (Ftot*Fin bf16)
    unsigned short* Qb  = (unsigned short*)(ws + 14 * MB);       // 8 MB  [B,H,N,64]
    unsigned short* Kb  = (unsigned short*)(ws + 22 * MB);       // 8 MB  [B,H,N,64]
    unsigned short* Vtb = (unsigned short*)(ws + 30 * MB);       // 8 MB  [B,H,64,N]

    conv_x_kernel<<<dim3((BATCH * NSEQ * FIN) / (256 * 4)), dim3(256), 0, stream>>>(x, xb);
    conv_wT_kernel<<<dim3((FIN * FTOT) / 256), dim3(256), 0, stream>>>(W, wt);

    qkv_gemm_kernel<<<dim3((BATCH * NSEQ) / 128, FTOT / 64), dim3(256), 0, stream>>>(
        xb, wt, Qb, Kb, Vtb);

    attn_kernel<<<dim3(NSEQ / 128, BATCH * NH), dim3(256), 0, stream>>>(Qb, Kb, Vtb, out);
}
